// EnhancedFastRCNN_19885698581227
// MI455X (gfx1250) — compile-verified, matrix operand order checked by reading, not run on hardware
//
#include <hip/hip_runtime.h>
#include <hip/hip_bf16.h>
#include <stdint.h>

// ---------------- problem constants ----------------
#define NB    32          // batch
#define NR    29          // regions
#define BRR   (NB*NR)     // 928 rows
#define NBOX  40
#define CCH   256         // channels
#define HF    64
#define WF    64
#define OUTP  7
#define D1    (CCH*OUTP*OUTP)  // 12544  (K of gemm1)
#define N1    2048             // hidden
#define N2    768              // out features
#define SCALE 0.0625f          // min(64/1024, 64/1024)

typedef __attribute__((ext_vector_type(16))) __bf16 v16bf;
typedef __attribute__((ext_vector_type(8)))  __bf16 bf16x8;
typedef __attribute__((ext_vector_type(4)))  __bf16 bf16x4;
typedef __attribute__((ext_vector_type(8)))  float  floatx8;
typedef __attribute__((ext_vector_type(2)))  int    int2v;
typedef __attribute__((ext_vector_type(4)))  int    int4v;

#define AS1 __attribute__((address_space(1)))
#define AS3 __attribute__((address_space(3)))

// CDNA5 async global->LDS path (ASYNCcnt-tracked), guarded by __has_builtin
#if defined(__has_builtin)
# if __has_builtin(__builtin_amdgcn_global_load_async_to_lds_b128) && \
     __has_builtin(__builtin_amdgcn_global_load_async_to_lds_b64) && \
     __has_builtin(__builtin_amdgcn_s_wait_asynccnt)
#  define USE_ASYNC 1
# endif
#endif
#ifndef USE_ASYNC
# define USE_ASYNC 0
#endif

// ---------------------------------------------------------------------------
// 1. Box selection: per (b,r) argmax of masked scores -> rois (feature coords)
// ---------------------------------------------------------------------------
__global__ void frcnn_select_kernel(const float* __restrict__ boxes,
                                    const float* __restrict__ scores,
                                    const int*   __restrict__ labels,
                                    float* __restrict__ rois,
                                    int*   __restrict__ det,
                                    float* __restrict__ detOut) {
    int t = threadIdx.x;                  // 0..927
    if (t >= BRR) return;
    int b = t / NR, r = t % NR;
    float best = -1e30f; int besti = 0; int found = 0;
    for (int n = 0; n < NBOX; ++n) {
        int lab = labels[b*NBOX + n];
        int ridx = (lab < 1 ? 1 : (lab > NR ? NR : lab)) - 1;
        int hit = (ridx == r);
        float mval = hit ? scores[b*NBOX + n] : -1.0f;
        if (mval > best) { best = mval; besti = n; }
        found |= hit;
    }
    const float* bx = boxes + ((size_t)b*NBOX + besti)*4;
    rois[t*4+0] = bx[0]*SCALE; rois[t*4+1] = bx[1]*SCALE;
    rois[t*4+2] = bx[2]*SCALE; rois[t*4+3] = bx[3]*SCALE;
    det[t] = found;
    detOut[t] = found ? 1.0f : 0.0f;
}

// ---------------------------------------------------------------------------
// 2. Transpose + fp32->bf16 convert:  dst[n*K+k] = src[k*N+n]
// ---------------------------------------------------------------------------
__global__ void frcnn_convT_kernel(const float* __restrict__ src,
                                   __bf16* __restrict__ dst, int K, int N) {
    size_t idx = (size_t)blockIdx.x * blockDim.x + threadIdx.x;
    if (idx >= (size_t)K * N) return;
    int n = (int)(idx / K);
    int k = (int)(idx % K);
    dst[idx] = (__bf16)src[(size_t)k * N + n];
}

// ---------------------------------------------------------------------------
// 3. ROI align (sampling_ratio=2, aligned=false) -> bf16 A matrix [928][12544]
// ---------------------------------------------------------------------------
__global__ void frcnn_roialign_kernel(const float* __restrict__ fm,
                                      const float* __restrict__ rois,
                                      __bf16* __restrict__ Abf) {
    __shared__ float sW[49*4*4];
    __shared__ int   sI[49*4*4];
    int br = blockIdx.x;
    int b  = br / NR;
    int tid = threadIdx.x;

    if (tid < 49*4) {
        int p = tid >> 2, samp = tid & 3;
        int py = p / OUTP, px = p % OUTP;
        int sy = samp >> 1, sx = samp & 1;
        float x1 = rois[br*4+0], y1 = rois[br*4+1];
        float x2 = rois[br*4+2], y2 = rois[br*4+3];
        float rw = fmaxf(x2 - x1, 1.0f), rh = fmaxf(y2 - y1, 1.0f);
        float bw = rw / OUTP, bh = rh / OUTP;
        float y = y1 + ((float)py + ((float)sy + 0.5f) * 0.5f) * bh;
        float x = x1 + ((float)px + ((float)sx + 0.5f) * 0.5f) * bw;
        float valid = (y >= -1.0f && y <= (float)HF && x >= -1.0f && x <= (float)WF) ? 1.0f : 0.0f;
        float yc = fminf(fmaxf(y, 0.0f), (float)(HF-1));
        float xc = fminf(fmaxf(x, 0.0f), (float)(WF-1));
        int y0 = (int)floorf(yc), x0 = (int)floorf(xc);
        int y1i = min(y0+1, HF-1), x1i = min(x0+1, WF-1);
        float ly = yc - (float)y0, lx = xc - (float)x0;
        float hy = 1.0f - ly, hx = 1.0f - lx;
        float s = 0.25f * valid;   // fold 1/(SR*SR) mean
        int base = tid*4;
        sW[base+0] = hy*hx*s; sW[base+1] = hy*lx*s;
        sW[base+2] = ly*hx*s; sW[base+3] = ly*lx*s;
        sI[base+0] = y0 *WF + x0;  sI[base+1] = y0 *WF + x1i;
        sI[base+2] = y1i*WF + x0;  sI[base+3] = y1i*WF + x1i;
    }
    __syncthreads();

    for (int e = tid; e < D1; e += blockDim.x) {
        int c = e / 49, p = e % 49;
        const float* fmc = fm + (((size_t)b * CCH + c) << 12);   // 64*64 plane
        float acc = 0.0f;
        #pragma unroll
        for (int s = 0; s < 4; ++s) {
            int si = (p*4 + s) * 4;
            acc += sW[si+0]*fmc[sI[si+0]] + sW[si+1]*fmc[sI[si+1]]
                 + sW[si+2]*fmc[sI[si+2]] + sW[si+3]*fmc[sI[si+3]];
        }
        Abf[(size_t)br * D1 + e] = (__bf16)acc;
    }
}

// ---------------------------------------------------------------------------
// 4. WMMA bf16 GEMM:  C[M,N] = A[M,K] * Bt[N,K]^T   (fp32 accumulate)
//    block = 256 threads = 8 wave32; C tile = 32 x 256; wave tile = 32 x 32
//    BK=32, triple-buffered LDS, async global->LDS staging, 1 barrier/iter
// ---------------------------------------------------------------------------
#define BKK  32
#define NBUF 3

__device__ __forceinline__ v16bf ld_frag(const __bf16* p) {
    bf16x8 lo = *(const bf16x8*)(p);        // K 0..7   (or 8..15 for hi lanes)
    bf16x8 hi = *(const bf16x8*)(p + 16);   // K 16..23 (or 24..31)
    return __builtin_shufflevector(lo, hi, 0,1,2,3,4,5,6,7,8,9,10,11,12,13,14,15);
}

__global__ void __launch_bounds__(256)
frcnn_wmma_gemm_kernel(const __bf16* __restrict__ A,
                       const __bf16* __restrict__ Bt,
                       float* __restrict__ C,
                       int M, int N, int K) {
    __shared__ __align__(16) __bf16 sA[NBUF][32  * BKK];   // 3 x 2KB
    __shared__ __align__(16) __bf16 sB[NBUF][256 * BKK];   // 3 x 16KB

    int tid   = threadIdx.x;
    int lane  = tid & 31;
    int wave  = tid >> 5;       // 0..7 : N sub-tile
    int mBlk  = blockIdx.y * 32;
    int nBlk  = blockIdx.x * 256;
    int fRow  = lane & 15;
    int kb    = (lane >> 4) * 8;   // 0 or 8 : K sub-offset per ISA frag layout

    floatx8 acc00 = {}, acc01 = {}, acc10 = {}, acc11 = {};

    // staging: A tile 32x32 bf16 (8B/thread), B tile 256x32 bf16 (64B/thread)
    int aRow = tid >> 3, aChunk = tid & 7;
    const __bf16* aSrc = A  + (size_t)(mBlk + aRow) * K + aChunk * 4;
    const __bf16* bSrc = Bt + (size_t)(nBlk + tid ) * K;

    int nIter = K / BKK;

    auto stage = [&](int it, int buf) {
        int kt = it * BKK;
#if USE_ASYNC
        __builtin_amdgcn_global_load_async_to_lds_b64(
            (int2v*)(aSrc + kt),
            (AS3 int2v*)(AS3 void*)(&sA[buf][aRow * BKK + aChunk * 4]), 0, 0);
        #pragma unroll
        for (int c = 0; c < 4; ++c)
            __builtin_amdgcn_global_load_async_to_lds_b128(
                (int4v*)(bSrc + kt + c * 8),
                (AS3 int4v*)(AS3 void*)(&sB[buf][tid * BKK + c * 8]), 0, 0);
#else
        *(bf16x4*)(&sA[buf][aRow * BKK + aChunk * 4]) = *(const bf16x4*)(aSrc + kt);
        #pragma unroll
        for (int c = 0; c < 4; ++c)
            *(bf16x8*)(&sB[buf][tid * BKK + c * 8]) = *(const bf16x8*)(bSrc + kt + c * 8);
#endif
    };

    stage(0, 0);

    for (int it = 0; it < nIter; ++it) {
        int buf = it % NBUF;
        bool hasNext = (it + 1) < nIter;
        if (hasNext) stage(it + 1, (it + 1) % NBUF);
#if USE_ASYNC
        // 5 async ops just issued for the next tile; async loads complete in
        // order, so <=5 outstanding guarantees the current tile has landed.
        if (hasNext) __builtin_amdgcn_s_wait_asynccnt(5);
        else         __builtin_amdgcn_s_wait_asynccnt(0);
#endif
        __syncthreads();

        v16bf a0 = ld_frag(&sA[buf][( 0 + fRow) * BKK + kb]);
        v16bf a1 = ld_frag(&sA[buf][(16 + fRow) * BKK + kb]);
        v16bf b0 = ld_frag(&sB[buf][(wave*32      + fRow) * BKK + kb]);
        v16bf b1 = ld_frag(&sB[buf][(wave*32 + 16 + fRow) * BKK + kb]);
        acc00 = __builtin_amdgcn_wmma_f32_16x16x32_bf16(false, a0, false, b0, (short)0, acc00, false, false);
        acc01 = __builtin_amdgcn_wmma_f32_16x16x32_bf16(false, a0, false, b1, (short)0, acc01, false, false);
        acc10 = __builtin_amdgcn_wmma_f32_16x16x32_bf16(false, a1, false, b0, (short)0, acc10, false, false);
        acc11 = __builtin_amdgcn_wmma_f32_16x16x32_bf16(false, a1, false, b1, (short)0, acc11, false, false);
    }

    // C layout: VGPR i, lanes 0-15 -> (M=i, N=lane); lanes 16-31 -> (M=8+i, N=lane-16)
    int mOut = mBlk + ((lane >> 4) << 3);
    int nOut = nBlk + wave * 32 + (lane & 15);
    #pragma unroll
    for (int i = 0; i < 8; ++i) {
        C[(size_t)(mOut + i     ) * N + nOut     ] = acc00[i];
        C[(size_t)(mOut + i     ) * N + nOut + 16] = acc01[i];
        C[(size_t)(mOut + i + 16) * N + nOut     ] = acc10[i];
        C[(size_t)(mOut + i + 16) * N + nOut + 16] = acc11[i];
    }
}

// ---------------------------------------------------------------------------
// 5. bias + LayerNorm + ReLU -> bf16 activations   (one block per row)
// ---------------------------------------------------------------------------
__global__ void frcnn_ln_relu_kernel(const float* __restrict__ X,
                                     const float* __restrict__ bias,
                                     const float* __restrict__ g,
                                     const float* __restrict__ be,
                                     __bf16* __restrict__ Y, int N) {
    __shared__ float r1[256], r2[256];
    int row = blockIdx.x, tid = threadIdx.x;
    const float* x = X + (size_t)row * N;
    float s = 0.f, s2 = 0.f;
    for (int j = tid; j < N; j += 256) {
        float v = x[j] + bias[j];
        s += v; s2 += v * v;
    }
    r1[tid] = s; r2[tid] = s2; __syncthreads();
    for (int off = 128; off > 0; off >>= 1) {
        if (tid < off) { r1[tid] += r1[tid+off]; r2[tid] += r2[tid+off]; }
        __syncthreads();
    }
    float mu = r1[0] / N;
    float var = r2[0] / N - mu * mu;
    float rstd = rsqrtf(var + 1e-5f);
    for (int j = tid; j < N; j += 256) {
        float v = (x[j] + bias[j] - mu) * rstd * g[j] + be[j];
        Y[(size_t)row * N + j] = (__bf16)fmaxf(v, 0.0f);
    }
}

// ---------------------------------------------------------------------------
// 6. bias + LayerNorm + missing-token select -> fp32 output
// ---------------------------------------------------------------------------
__global__ void frcnn_final_kernel(const float* __restrict__ X,
                                   const float* __restrict__ bias,
                                   const float* __restrict__ g,
                                   const float* __restrict__ be,
                                   const float* __restrict__ missing,
                                   const int*   __restrict__ det,
                                   float* __restrict__ out, int N) {
    __shared__ float r1[256], r2[256];
    int row = blockIdx.x, tid = threadIdx.x;
    const float* x = X + (size_t)row * N;
    float s = 0.f, s2 = 0.f;
    for (int j = tid; j < N; j += 256) {
        float v = x[j] + bias[j];
        s += v; s2 += v * v;
    }
    r1[tid] = s; r2[tid] = s2; __syncthreads();
    for (int off = 128; off > 0; off >>= 1) {
        if (tid < off) { r1[tid] += r1[tid+off]; r2[tid] += r2[tid+off]; }
        __syncthreads();
    }
    float mu = r1[0] / N;
    float var = r2[0] / N - mu * mu;
    float rstd = rsqrtf(var + 1e-5f);
    int d = det[row];
    for (int j = tid; j < N; j += 256) {
        float v = (x[j] + bias[j] - mu) * rstd * g[j] + be[j];
        out[(size_t)row * N + j] = d ? v : missing[j];
    }
}

// ---------------------------------------------------------------------------
extern "C" void kernel_launch(void* const* d_in, const int* in_sizes, int n_in,
                              void* d_out, int out_size, void* d_ws, size_t ws_size,
                              hipStream_t stream) {
    const float* fm      = (const float*)d_in[0];
    const float* boxes   = (const float*)d_in[1];
    const float* scores  = (const float*)d_in[2];
    const int*   labels  = (const int*)  d_in[3];
    const float* w1      = (const float*)d_in[4];
    const float* b1      = (const float*)d_in[5];
    const float* g1      = (const float*)d_in[6];
    const float* be1     = (const float*)d_in[7];
    const float* w2      = (const float*)d_in[8];
    const float* b2      = (const float*)d_in[9];
    const float* g2      = (const float*)d_in[10];
    const float* be2     = (const float*)d_in[11];
    const float* missing = (const float*)d_in[12];
    float* out = (float*)d_out;

    // workspace carve (256B aligned)
    uint8_t* p = (uint8_t*)d_ws;
    auto carve = [&](size_t bytes) -> uint8_t* {
        uint8_t* r = p;
        p += (bytes + 255) & ~(size_t)255;
        return r;
    };
    __bf16* w1t = (__bf16*)carve((size_t)N1 * D1 * 2);   // [2048][12544]
    __bf16* w2t = (__bf16*)carve((size_t)N2 * N1 * 2);   // [768][2048]
    __bf16* Abf = (__bf16*)carve((size_t)BRR * D1 * 2);  // [928][12544]
    float*  h1  = (float*) carve((size_t)BRR * N1 * 4);  // [928][2048]
    __bf16* h1b = (__bf16*)carve((size_t)BRR * N1 * 2);
    float*  h2  = (float*) carve((size_t)BRR * N2 * 4);  // [928][768]
    float*  rois= (float*) carve((size_t)BRR * 4 * 4);
    int*    det = (int*)   carve((size_t)BRR * 4);

    float* detOut = out + (size_t)BRR * N2;              // detected tail of d_out

    // 1. box selection
    frcnn_select_kernel<<<1, BRR, 0, stream>>>(boxes, scores, labels, rois, det, detOut);

    // 2. weight transpose-converts
    {
        size_t n = (size_t)N1 * D1;
        frcnn_convT_kernel<<<(unsigned)((n + 255) / 256), 256, 0, stream>>>(w1, w1t, D1, N1);
    }
    {
        size_t n = (size_t)N2 * N1;
        frcnn_convT_kernel<<<(unsigned)((n + 255) / 256), 256, 0, stream>>>(w2, w2t, N1, N2);
    }

    // 3. ROI align -> bf16 A
    frcnn_roialign_kernel<<<BRR, 256, 0, stream>>>(fm, rois, Abf);

    // 4. GEMM1: [928 x 12544] x [12544 x 2048]
    frcnn_wmma_gemm_kernel<<<dim3(N1/256, BRR/32), 256, 0, stream>>>(Abf, w1t, h1, BRR, N1, D1);

    // 5. LN + ReLU -> bf16
    frcnn_ln_relu_kernel<<<BRR, 256, 0, stream>>>(h1, b1, g1, be1, h1b, N1);

    // 6. GEMM2: [928 x 2048] x [2048 x 768]
    frcnn_wmma_gemm_kernel<<<dim3(N2/256, BRR/32), 256, 0, stream>>>(h1b, w2t, h2, BRR, N2, N1);

    // 7. LN + missing-token select -> fp32 out
    frcnn_final_kernel<<<BRR, 256, 0, stream>>>(h2, b2, g2, be2, missing, det, out, N2);
}
